// Encoder_81303730913792
// MI455X (gfx1250) — matrix operand order repaired and checked by reading, not compiled, not run on hardware
//
#include <hip/hip_runtime.h>
#include <hip/hip_bf16.h>

#define B_  32
#define S_  512
#define D_  512
#define H_  16
#define K_  32
#define A_  256
#define HK_ 512

typedef __bf16 bf16_t;
typedef __attribute__((ext_vector_type(16))) __bf16 v16bf;
typedef __attribute__((ext_vector_type(8)))  __bf16 v8bf;
typedef __attribute__((ext_vector_type(8)))  float  v8f;

// ---------------- WMMA fragment helpers (CDNA5 16x16x32 bf16) ----------------

static __device__ __forceinline__ v16bf frag_a_bf16(const bf16_t* base, int ld,
                                                    int r0, int k0, int lane) {
  const int m = lane & 15, lh = lane >> 4;
  const bf16_t* row = base + (size_t)(r0 + m) * ld + k0;
  v8bf lo = *(const v8bf*)(row + lh * 8);
  v8bf hi = *(const v8bf*)(row + 16 + lh * 8);
  v16bf f;
#pragma unroll
  for (int i = 0; i < 8; ++i) { f[i] = lo[i]; f[i + 8] = hi[i]; }
  return f;
}

static __device__ __forceinline__ v16bf frag_a_f32(const float* base, int ld,
                                                   int r0, int k0, int lane) {
  const int m = lane & 15, lh = lane >> 4;
  const float* row = base + (size_t)(r0 + m) * ld + k0;
  v16bf f;
#pragma unroll
  for (int i = 0; i < 8; ++i) {
    f[i]     = (bf16_t)row[lh * 8 + i];
    f[i + 8] = (bf16_t)row[16 + lh * 8 + i];
  }
  return f;
}

// Source is "Bsrc[n][k]" (B transposed, row-major, contiguous in k).
static __device__ __forceinline__ v16bf frag_b_bf16(const bf16_t* base, int ld,
                                                    int n0, int k0, int lane) {
  const int n = lane & 15, lh = lane >> 4;
  const bf16_t* row = base + (size_t)(n0 + n) * ld + k0;
  v8bf lo = *(const v8bf*)(row + lh * 8);
  v8bf hi = *(const v8bf*)(row + 16 + lh * 8);
  v16bf f;
#pragma unroll
  for (int i = 0; i < 8; ++i) { f[i] = lo[i]; f[i + 8] = hi[i]; }
  return f;
}

static __device__ __forceinline__ v8f wmma_bf16(v16bf a, v16bf b, v8f c) {
  return __builtin_amdgcn_wmma_f32_16x16x32_bf16(false, a, false, b,
                                                 (short)0, c, false, false);
}

// ---------------- convert / transpose kernels --------------------------------

__global__ __launch_bounds__(256) void cvt_x(const float* __restrict__ x,
                                             bf16_t* __restrict__ xbf,
                                             bf16_t* __restrict__ xtbf) {
  size_t i = (size_t)blockIdx.x * 256 + threadIdx.x;   // B*S*D
  int d = (int)(i % D_);
  size_t bs = i / D_;
  int s = (int)(bs % S_);
  int b = (int)(bs / S_);
  float v = x[i];
  xbf[i] = (bf16_t)v;
  xtbf[((size_t)b * D_ + d) * S_ + s] = (bf16_t)v;
}

__global__ __launch_bounds__(256) void cvt_qw(const float* __restrict__ Qw,
                                              bf16_t* __restrict__ qwt) {
  size_t i = (size_t)blockIdx.x * 256 + threadIdx.x;   // H*D*D
  int e = (int)(i % D_);
  size_t hd = i / D_;
  int d = (int)(hd % D_);
  int h = (int)(hd / D_);
  qwt[((size_t)h * D_ + e) * D_ + d] = (bf16_t)Qw[i];
}

__global__ __launch_bounds__(256) void cvt_vw(const float* __restrict__ Vw,
                                              bf16_t* __restrict__ vwt) {
  size_t i = (size_t)blockIdx.x * 256 + threadIdx.x;   // H*D*K
  int k = (int)(i % K_);
  size_t hd = i / K_;
  int d = (int)(hd % D_);
  int h = (int)(hd / D_);
  vwt[((size_t)h * K_ + k) * D_ + d] = (bf16_t)Vw[i];
}

__global__ __launch_bounds__(256) void cvt_wv(const float* __restrict__ Wv,
                                              bf16_t* __restrict__ wvt) {
  size_t i = (size_t)blockIdx.x * 256 + threadIdx.x;   // HK*A
  int a = (int)(i % A_);
  int j = (int)(i / A_);
  wvt[(size_t)a * HK_ + j] = (bf16_t)Wv[i];
}

// ---------------- kernel 1: q-tile + raw scores ------------------------------
// grid = B*H*(S/TQ); one 32-row query tile per workgroup.

#define TQ  32
#define LDP 520

__global__ __launch_bounds__(256)
void qk_kernel(const float* __restrict__ Qb,
               const bf16_t* __restrict__ xbf, const bf16_t* __restrict__ qwt,
               bf16_t* __restrict__ scores) {
  const int blk = blockIdx.x;
  const int bh = blk >> 4, qt = blk & 15;
  const int b = bh / H_, h = bh % H_;
  const int tid  = threadIdx.x;
  const int wave = tid >> 5, lane = tid & 31;
  const int n16 = lane & 15, lh = lane >> 4;

  __shared__ __align__(16) bf16_t tlds[TQ * LDP];

  const bf16_t* xb = xbf + (size_t)b * S_ * D_;
  const bf16_t* qw = qwt + (size_t)h * D_ * D_;
  bf16_t* sc = scores + (size_t)bh * S_ * S_;

  const int rt    = wave & 1;
  const int cbase = (wave >> 1) * 8;
  const int r0 = qt * TQ;

  // GEMM1: q = x Qw^T + Qb  (M=32, N=512, K=512) -> LDS
  {
    v8f acc[8];
#pragma unroll
    for (int c = 0; c < 8; ++c) acc[c] = (v8f){};
    for (int k0 = 0; k0 < D_; k0 += 32) {
      v16bf af = frag_a_bf16(xb, D_, r0 + rt * 16, k0, lane);
#pragma unroll
      for (int c = 0; c < 8; ++c) {
        v16bf bfrag = frag_b_bf16(qw, D_, (cbase + c) * 16, k0, lane);
        acc[c] = wmma_bf16(af, bfrag, acc[c]);
      }
    }
#pragma unroll
    for (int c = 0; c < 8; ++c) {
      const int col = (cbase + c) * 16 + n16;
      const float qbv = Qb[h * D_ + col];
#pragma unroll
      for (int i = 0; i < 8; ++i)
        tlds[(rt * 16 + lh * 8 + i) * LDP + col] = (bf16_t)(acc[c][i] + qbv);
    }
  }
  __syncthreads();

  // GEMM2: scores = q x^T  (M=32, N=512, K=512) -> global scratch
  {
    v8f acc[8];
#pragma unroll
    for (int c = 0; c < 8; ++c) acc[c] = (v8f){};
    for (int k0 = 0; k0 < D_; k0 += 32) {
      v16bf af = frag_a_bf16(tlds, LDP, rt * 16, k0, lane);
#pragma unroll
      for (int c = 0; c < 8; ++c) {
        v16bf bfrag = frag_b_bf16(xb, D_, (cbase + c) * 16, k0, lane);
        acc[c] = wmma_bf16(af, bfrag, acc[c]);
      }
    }
#pragma unroll
    for (int c = 0; c < 8; ++c) {
      const int t = (cbase + c) * 16 + n16;
#pragma unroll
      for (int i = 0; i < 8; ++i) {
        int sq = r0 + rt * 16 + lh * 8 + i;
        sc[(size_t)sq * S_ + t] = (bf16_t)acc[c][i];
      }
    }
  }
}

// ---------------- kernel 2a: per-column softmax stats ------------------------
// one thread per (bh, t) column; coalesced row reads across lanes.

__global__ __launch_bounds__(256)
void stats_kernel(const bf16_t* __restrict__ scores,
                  float* __restrict__ Mr, float* __restrict__ Rr) {
  const int gid = blockIdx.x * 256 + threadIdx.x;   // B*H*S columns
  const int bh = gid >> 9;
  const int t  = gid & (S_ - 1);
  const bf16_t* sc = scores + (size_t)bh * S_ * S_;
  float m = -3.0e38f;
  for (int s = 0; s < S_; ++s)
    m = fmaxf(m, (float)sc[(size_t)s * S_ + t]);
  float z = 0.0f;
  for (int s = 0; s < S_; ++s)
    z += __expf((float)sc[(size_t)s * S_ + t] - m);
  Mr[gid] = m;
  Rr[gid] = 1.0f / z;
}

// ---------------- kernel 2b: normalize scores in place (P, bf16) -------------

__global__ __launch_bounds__(256)
void norm_kernel(bf16_t* __restrict__ scores,
                 const float* __restrict__ Mr, const float* __restrict__ Rr) {
  const size_t idx = ((size_t)blockIdx.x * 256 + threadIdx.x) * 16;
  const int bh = (int)(idx >> 18);                  // S*S = 2^18
  const int off = (int)(idx & (size_t)(S_ * S_ - 1));
  const int t0 = off & (S_ - 1);
  const float* M = Mr + bh * S_;
  const float* R = Rr + bh * S_;
  bf16_t* p = scores + idx;
  v8bf v0 = *(const v8bf*)(p);
  v8bf v1 = *(const v8bf*)(p + 8);
#pragma unroll
  for (int j = 0; j < 8; ++j) {
    v0[j] = (bf16_t)(__expf((float)v0[j] - M[t0 + j])     * R[t0 + j]);
    v1[j] = (bf16_t)(__expf((float)v1[j] - M[t0 + 8 + j]) * R[t0 + 8 + j]);
  }
  *(v8bf*)(p)     = v0;
  *(v8bf*)(p + 8) = v1;
}

// ---------------- kernel 3: context + per-head V projection ------------------
// grid = B*H*(S/TQ).

__global__ __launch_bounds__(256)
void ctx_kernel(const float* __restrict__ Vb,
                const bf16_t* __restrict__ xtbf, const bf16_t* __restrict__ vwt,
                const bf16_t* __restrict__ scores, float* __restrict__ hout) {
  const int blk = blockIdx.x;
  const int bh = blk >> 4, qt = blk & 15;
  const int b = bh / H_, h = bh % H_;
  const int tid  = threadIdx.x;
  const int wave = tid >> 5, lane = tid & 31;
  const int n16 = lane & 15, lh = lane >> 4;

  __shared__ __align__(16) bf16_t tlds[TQ * LDP];

  const bf16_t* xtb = xtbf + (size_t)b * D_ * S_;
  const bf16_t* vw  = vwt  + (size_t)h * K_ * D_;
  const bf16_t* sc  = scores + (size_t)bh * S_ * S_;

  const int rt    = wave & 1;
  const int cbase = (wave >> 1) * 8;
  const int r0 = qt * TQ;

  // GEMM3: ctx = P x  (M=32, N=512, K=512) -> LDS
  {
    v8f acc[8];
#pragma unroll
    for (int c = 0; c < 8; ++c) acc[c] = (v8f){};
    for (int k0 = 0; k0 < S_; k0 += 32) {
      v16bf af = frag_a_bf16(sc, S_, r0 + rt * 16, k0, lane);
#pragma unroll
      for (int c = 0; c < 8; ++c) {
        v16bf bfrag = frag_b_bf16(xtb, S_, (cbase + c) * 16, k0, lane);
        acc[c] = wmma_bf16(af, bfrag, acc[c]);
      }
    }
#pragma unroll
    for (int c = 0; c < 8; ++c) {
      const int col = (cbase + c) * 16 + n16;
#pragma unroll
      for (int i = 0; i < 8; ++i)
        tlds[(rt * 16 + lh * 8 + i) * LDP + col] = (bf16_t)acc[c][i];
    }
  }
  __syncthreads();

  // GEMM4: head_out = ctx Vw^T + Vb  (M=32, N=32, K=512)
  if (wave < 4) {
    const int rt4 = wave & 1, ct4 = wave >> 1;
    v8f acc = {};
    for (int k0 = 0; k0 < D_; k0 += 32) {
      v16bf af = frag_a_bf16(tlds, LDP, rt4 * 16, k0, lane);
      v16bf bfrag = frag_b_bf16(vw, D_, ct4 * 16, k0, lane);
      acc = wmma_bf16(af, bfrag, acc);
    }
    const int kc = ct4 * 16 + n16;
    const float vb = Vb[h * K_ + kc];
#pragma unroll
    for (int i = 0; i < 8; ++i) {
      int sq = r0 + rt4 * 16 + lh * 8 + i;
      hout[((size_t)(b * S_ + sq)) * HK_ + h * K_ + kc] = acc[i] + vb;
    }
  }
}

// ---------------- kernel 4: additive-attention pooling -----------------------

__global__ __launch_bounds__(256)
void pool_kernel(const float* __restrict__ bv, const float* __restrict__ wq,
                 const float* __restrict__ bq, const bf16_t* __restrict__ wvt,
                 const float* __restrict__ hout, float* __restrict__ out) {
  const int b = blockIdx.x;
  const int tid  = threadIdx.x;
  const int wave = tid >> 5, lane = tid & 31;
  const int n16 = lane & 15, lh = lane >> 4;

  __shared__ float a_lds[S_];
  for (int s = tid; s < S_; s += 256) a_lds[s] = 0.0f;
  __syncthreads();

  const float* hb = hout + (size_t)b * S_ * HK_;

  // t = tanh(h Wv + bv);  a[s] += t[s,:] . wq   (M=512, N=256, K=512)
  for (int tile = wave; tile < 32 * 16; tile += 8) {
    const int rtile = tile >> 4, ctile = tile & 15;
    v8f acc = {};
    for (int k0 = 0; k0 < HK_; k0 += 32) {
      v16bf af = frag_a_f32(hb, HK_, rtile * 16, k0, lane);
      v16bf bfrag = frag_b_bf16(wvt, HK_, ctile * 16, k0, lane);
      acc = wmma_bf16(af, bfrag, acc);
    }
    const int col = ctile * 16 + n16;
    const float bvc = bv[col], wqc = wq[col];
#pragma unroll
    for (int i = 0; i < 8; ++i) {
      int row = rtile * 16 + lh * 8 + i;
      atomicAdd(&a_lds[row], tanhf(acc[i] + bvc) * wqc);
    }
  }
  __syncthreads();
  for (int s = tid; s < S_; s += 256) a_lds[s] += bq[0];
  __syncthreads();

  for (int k = tid; k < HK_; k += 256) {
    float z = 0.0f;
    for (int s = 0; s < S_; ++s)
      z += a_lds[s] * hb[(size_t)s * HK_ + k];
    out[b * HK_ + k] = z;
  }
}

// ---------------- launcher ---------------------------------------------------

extern "C" void kernel_launch(void* const* d_in, const int* in_sizes, int n_in,
                              void* d_out, int out_size, void* d_ws, size_t ws_size,
                              hipStream_t stream) {
  const float* x  = (const float*)d_in[0];
  const float* Qw = (const float*)d_in[1];
  const float* Qb = (const float*)d_in[2];
  const float* Vw = (const float*)d_in[3];
  const float* Vb = (const float*)d_in[4];
  const float* Wv = (const float*)d_in[5];
  const float* bv = (const float*)d_in[6];
  const float* wq = (const float*)d_in[7];
  const float* bq = (const float*)d_in[8];
  float* out = (float*)d_out;

  char* p = (char*)d_ws;
  bf16_t* xbf  = (bf16_t*)p; p += (size_t)B_ * S_ * D_ * 2;      // 16 MB
  bf16_t* xtbf = (bf16_t*)p; p += (size_t)B_ * S_ * D_ * 2;      // 16 MB
  bf16_t* qwt  = (bf16_t*)p; p += (size_t)H_ * D_ * D_ * 2;      // 8 MB
  bf16_t* vwt  = (bf16_t*)p; p += (size_t)H_ * K_ * D_ * 2;      // 0.5 MB
  bf16_t* wvt  = (bf16_t*)p; p += (size_t)A_ * HK_ * 2;          // 0.25 MB
  float*  Mr   = (float*)p;  p += (size_t)B_ * H_ * S_ * 4;      // 1 MB
  float*  Rr   = (float*)p;  p += (size_t)B_ * H_ * S_ * 4;      // 1 MB
  float*  hout = (float*)p;  p += (size_t)B_ * S_ * HK_ * 4;     // 32 MB
  bf16_t* scor = (bf16_t*)p; p += (size_t)B_ * H_ * S_ * S_ * 2; // 256 MB

  cvt_x <<<(B_ * S_ * D_) / 256, 256, 0, stream>>>(x, xbf, xtbf);
  cvt_qw<<<(H_ * D_ * D_) / 256, 256, 0, stream>>>(Qw, qwt);
  cvt_vw<<<(H_ * D_ * K_) / 256, 256, 0, stream>>>(Vw, vwt);
  cvt_wv<<<(HK_ * A_)     / 256, 256, 0, stream>>>(Wv, wvt);

  qk_kernel   <<<B_ * H_ * (S_ / TQ), 256, 0, stream>>>(Qb, xbf, qwt, scor);
  stats_kernel<<<(B_ * H_ * S_) / 256, 256, 0, stream>>>(scor, Mr, Rr);
  norm_kernel <<<(int)(((size_t)B_ * H_ * S_ * S_) / (256 * 16)), 256, 0, stream>>>(scor, Mr, Rr);
  ctx_kernel  <<<B_ * H_ * (S_ / TQ), 256, 0, stream>>>(Vb, xtbf, vwt, scor, hout);
  pool_kernel <<<B_, 256, 0, stream>>>(bv, wq, bq, wvt, hout, out);

  (void)in_sizes; (void)n_in; (void)out_size; (void)ws_size;
}
